// SRFormer3D_9964324126846
// MI455X (gfx1250) — compile-verified, matrix operand order checked by reading, not run on hardware
//
#include <hip/hip_runtime.h>
#include <hip/hip_bf16.h>
#include <cstdint>
#include <cstddef>

// ---------------------------------------------------------------------------
// SRFormer3D fused attention for gfx1250 (MI455X, wave32, WMMA).
// All matmuls use v_wmma_f32_16x16x32_bf16 (f32 accumulate).
// Pipeline: prep (weightsT->bf16, bias table)
//        -> fused q+kv projection GEMM (x read once)
//        -> fused attention+output-projection (per 64-query chunk: 6 heads of
//           QK + softmax + PV staged in LDS, then 64x192 @ 192x192 proj GEMM
//           straight from LDS to f32 d_out).  No attn-out HBM round-trip.
// ---------------------------------------------------------------------------

typedef __attribute__((ext_vector_type(16))) __bf16 bf16x16;
typedef __attribute__((ext_vector_type(2)))  __bf16 bf16x2;
typedef __attribute__((ext_vector_type(8)))  float  f32x8;

union BFVec { unsigned short u[16]; bf16x16 v; };

static __device__ __forceinline__ unsigned short f2bf(float f) {
  unsigned int u = __float_as_uint(f);
  u += 0x7FFFu + ((u >> 16) & 1u);          // round-to-nearest-even
  return (unsigned short)(u >> 16);
}

// Convert 16 f32 -> bf16x16 (packed cvt if the target declares it).
static __device__ __forceinline__ bf16x16 cvt16(const float* fv) {
#if __has_builtin(__builtin_amdgcn_cvt_pk_bf16_f32)
  union { bf16x2 p[8]; bf16x16 v; } r;
  #pragma unroll
  for (int j = 0; j < 8; ++j)
    r.p[j] = __builtin_amdgcn_cvt_pk_bf16_f32(fv[2 * j], fv[2 * j + 1]);
  return r.v;
#else
  BFVec a;
  #pragma unroll
  for (int s = 0; s < 16; ++s) a.u[s] = f2bf(fv[s]);
  return a.v;
#endif
}

static __device__ __forceinline__ f32x8 wmma_bf16(bf16x16 a, bf16x16 b, f32x8 c) {
  // (neg_a, A, neg_b, B, c_mod, C, reuse_a, reuse_b)
  return __builtin_amdgcn_wmma_f32_16x16x32_bf16(false, a, false, b, (short)0, c,
                                                 false, false);
}

static constexpr int kB   = 512;   // windows
static constexpr int kN   = 512;   // queries / window
static constexpr int kC   = 192;   // channels
static constexpr int kNH  = 6;     // heads
static constexpr int kHD  = 32;    // head dim
static constexpr int kNK  = 64;    // kv tokens / window
static constexpr float kScale = 0.17677669529663689f; // 32^-0.5

// ---------------------------------------------------------------------------
// Prep: transpose + bf16-convert weights so B-fragments (Wt[n][k]) are
// contiguous 32B loads.
// ---------------------------------------------------------------------------
__global__ void prep_weights(const float* __restrict__ qw,
                             const float* __restrict__ kvw,
                             const float* __restrict__ pw,
                             unsigned short* __restrict__ qWt,
                             unsigned short* __restrict__ kvWt,
                             unsigned short* __restrict__ pWt) {
  int i = blockIdx.x * blockDim.x + threadIdx.x;
  if (i < kC * kC) {
    int n = i / kC, k = i % kC;
    qWt[n * kC + k] = f2bf(qw[k * kC + n]);
  }
  int j = i - kC * kC;
  if (j >= 0 && j < 48 * kC) {
    int n = j / kC, k = j % kC;
    kvWt[n * kC + k] = f2bf(kvw[k * 48 + n]);
  }
  int l = i - kC * kC - 48 * kC;
  if (l >= 0 && l < kC * kC) {
    int n = l / kC, k = l % kC;
    pWt[n * kC + k] = f2bf(pw[k * kC + n]);
  }
}

// ---------------------------------------------------------------------------
// Prep: relative-position bias[h][n(512)][t(64)] from the 343x6 table.
// Index math recomputed on device (host-side _ALIGNED_IDX is not an input).
// ---------------------------------------------------------------------------
__global__ void prep_bias(const float* __restrict__ rpb,
                          float* __restrict__ bias) {
  int i = blockIdx.x * blockDim.x + threadIdx.x;
  if (i >= kNH * kN * kNK) return;
  int t = i & 63;
  int n = (i >> 6) & 511;
  int h = i >> 15;
  int i2 = n & 7, i1 = (n >> 3) & 7, i0 = n >> 6;       // query coord in 8^3
  int t2 = t & 3, t1 = (t >> 2) & 3, t0 = (t >> 4) & 3; // kv coord in 4^3
  int d0 = (i0 >> 1) - t0 + 3;
  int d1 = (i1 >> 1) - t1 + 3;
  int d2 = (i2 >> 1) - t2 + 3;
  int idx = (d0 * 7 + d1) * 7 + d2;                     // 7^3 = 343 table
  bias[i] = rpb[idx * kNH + h];
}

// ---------------------------------------------------------------------------
// Fused GEMM: q = (x @ q_w + q_b) * scale  and  kv = x @ kv_w + kv_b.
// x is read ONCE.  256 threads = 8 waves; each wave owns a 16-row strip and
// 15 N-subtiles (12 q -> qbf, 3 kv -> permuted scatter into k / vT).
// ---------------------------------------------------------------------------
__global__ __launch_bounds__(256) void gemm_qkv(const float* __restrict__ x,
                                                const unsigned short* __restrict__ qWt,
                                                const unsigned short* __restrict__ kvWt,
                                                const float* __restrict__ qb,
                                                const float* __restrict__ kvb,
                                                unsigned short* __restrict__ qbf,
                                                unsigned short* __restrict__ kbf,
                                                unsigned short* __restrict__ vT) {
  const int lane = threadIdx.x & 31;
  const int hf   = lane >> 4;
  const int l15  = lane & 15;
  const int mbase = blockIdx.x * 128 + (threadIdx.x >> 5) * 16;
  const int mrow  = mbase + l15;

  f32x8 qacc[12];
  f32x8 kacc[3];
  #pragma unroll
  for (int n = 0; n < 12; ++n)
    #pragma unroll
    for (int s = 0; s < 8; ++s) qacc[n][s] = 0.f;
  #pragma unroll
  for (int n = 0; n < 3; ++n)
    #pragma unroll
    for (int s = 0; s < 8; ++s) kacc[n][s] = 0.f;

  #pragma unroll
  for (int kb = 0; kb < kC; kb += 32) {
    // A fragment: 16 consecutive f32 -> bf16 (half picks K 0..15 / 16..31)
    const float4* ap = reinterpret_cast<const float4*>(
        x + (size_t)mrow * kC + kb + hf * 16);
    float fv[16];
    #pragma unroll
    for (int j = 0; j < 4; ++j) {
      float4 f = ap[j];
      fv[4 * j + 0] = f.x; fv[4 * j + 1] = f.y;
      fv[4 * j + 2] = f.z; fv[4 * j + 3] = f.w;
    }
    bf16x16 afrag = cvt16(fv);

    #pragma unroll
    for (int n = 0; n < 12; ++n) {
      bf16x16 bfrag = *reinterpret_cast<const bf16x16*>(
          qWt + (size_t)(n * 16 + l15) * kC + kb + hf * 16);
      qacc[n] = wmma_bf16(afrag, bfrag, qacc[n]);
    }
    #pragma unroll
    for (int n = 0; n < 3; ++n) {
      bf16x16 bfrag = *reinterpret_cast<const bf16x16*>(
          kvWt + (size_t)(n * 16 + l15) * kC + kb + hf * 16);
      kacc[n] = wmma_bf16(afrag, bfrag, kacc[n]);
    }
  }

  // ---- q epilogue: +bias, *scale, bf16 store [B*N, 192] ------------------
  #pragma unroll
  for (int n = 0; n < 12; ++n) {
    int col = n * 16 + l15;
    float bb = qb[col];
    #pragma unroll
    for (int r = 0; r < 8; ++r) {
      int row = mbase + r + 8 * hf;
      qbf[(size_t)row * kC + col] = f2bf((qacc[n][r] + bb) * kScale);
    }
  }

  // ---- kv epilogue: 2x2x2 sub-position permutation scatter ---------------
  #pragma unroll
  for (int n = 0; n < 3; ++n) {
    int col   = n * 16 + l15;          // 0..47
    float bb  = kvb[col];
    int kvsel = col / 24;              // 0 = k, 1 = v
    int h     = (col >> 2) % kNH;
    int cc    = col & 3;
    #pragma unroll
    for (int r = 0; r < 8; ++r) {
      int row = mbase + r + 8 * hf;    // global B*N row
      int b_  = row >> 9;
      int nn  = row & 511;
      int i2 = nn & 7, i1 = (nn >> 3) & 7, i0 = nn >> 6;
      int t  = ((i0 >> 1) * 4 + (i1 >> 1)) * 4 + (i2 >> 1);         // 0..63
      int dd = (((i0 & 1) * 2 + (i1 & 1)) * 2 + (i2 & 1)) * 4 + cc; // 0..31
      unsigned short val = f2bf(kacc[n][r] + bb);
      if (kvsel == 0)
        kbf[(((size_t)b_ * kNH + h) * kNK + t) * kHD + dd] = val;
      else
        vT[(((size_t)b_ * kNH + h) * kHD + dd) * kNK + t] = val;
    }
  }
}

// ---------------------------------------------------------------------------
// Fused attention + output projection.
// Block = (window b, 64-query chunk), 128 threads / 4 waves.
// For each head: QK (4 WMMA/wave) -> +bias -> LDS softmax -> PV (4 WMMA/wave)
// parked as bf16 in s_ao[64][192].  Then proj GEMM (72 WMMA/wave) from LDS,
// writing f32 straight to d_out.  Removes the 200 MB attn-out HBM round-trip.
// ---------------------------------------------------------------------------
__global__ __launch_bounds__(128) void attn_proj(const unsigned short* __restrict__ qbf,
                                                 const unsigned short* __restrict__ kbf,
                                                 const unsigned short* __restrict__ vT,
                                                 const float* __restrict__ bias,
                                                 const unsigned short* __restrict__ pWt,
                                                 const float* __restrict__ pb,
                                                 float* __restrict__ out) {
  __shared__ alignas(64) float          s_attn[64 * 64];  // scores (f32), 16 KB
  __shared__ alignas(64) unsigned short s_p[64 * 64];     // probs (bf16),  8 KB
  __shared__ alignas(64) unsigned short s_ao[64 * kC];    // attn out,     24 KB

  const int qc = blockIdx.x & 7;
  const int b  = blockIdx.x >> 3;
  const int w    = threadIdx.x >> 5;
  const int lane = threadIdx.x & 31;
  const int hf   = lane >> 4;
  const int l15  = lane & 15;
  const int q0   = qc * 64;

  for (int h = 0; h < kNH; ++h) {
    // ---- QK^T : 16x64 scores per wave (K = 32 = one WMMA step) -----------
    bf16x16 aq = *reinterpret_cast<const bf16x16*>(
        qbf + ((size_t)b * kN + q0 + w * 16 + l15) * kC + h * kHD + hf * 16);
    const unsigned short* kp = kbf + ((size_t)b * kNH + h) * kNK * kHD;

    f32x8 acc[4];
    #pragma unroll
    for (int a = 0; a < 4; ++a) {
      f32x8 c;
      #pragma unroll
      for (int s = 0; s < 8; ++s) c[s] = 0.f;
      bf16x16 bk = *reinterpret_cast<const bf16x16*>(
          kp + (size_t)(a * 16 + l15) * kHD + hf * 16);
      acc[a] = wmma_bf16(aq, bk, c);
    }

    // ---- bias add, spill to LDS (also guards prev head's s_p reads) ------
    __syncthreads();
    const float* bp = bias + ((size_t)h * kN + q0 + w * 16) * kNK;
    #pragma unroll
    for (int a = 0; a < 4; ++a)
      #pragma unroll
      for (int r = 0; r < 8; ++r) {
        int m = r + 8 * hf;
        int t = a * 16 + l15;
        s_attn[(w * 16 + m) * 64 + t] = acc[a][r] + bp[m * 64 + t];
      }
    __syncthreads();

    // ---- softmax over 64 kv tokens (one thread per query row) ------------
    if (threadIdx.x < 64) {
      float* row = s_attn + threadIdx.x * 64;
      float mx = -3.0e38f;
      #pragma unroll 4
      for (int t = 0; t < 64; ++t) mx = fmaxf(mx, row[t]);
      float sum = 0.f;
      #pragma unroll 4
      for (int t = 0; t < 64; ++t) sum += __expf(row[t] - mx);
      float inv = 1.f / sum;
      #pragma unroll 4
      for (int t = 0; t < 64; ++t)
        s_p[threadIdx.x * 64 + t] = f2bf(__expf(row[t] - mx) * inv);
    }
    __syncthreads();

    // ---- P @ V : 16x32 per wave (K = 64 = two WMMA steps) -> s_ao --------
    const unsigned short* vp = vT + ((size_t)b * kNH + h) * kHD * kNK;
    #pragma unroll
    for (int nn = 0; nn < 2; ++nn) {
      f32x8 c;
      #pragma unroll
      for (int s = 0; s < 8; ++s) c[s] = 0.f;
      #pragma unroll
      for (int k2 = 0; k2 < 2; ++k2) {
        bf16x16 ap = *reinterpret_cast<const bf16x16*>(
            s_p + (w * 16 + l15) * 64 + k2 * 32 + hf * 16);
        bf16x16 bv = *reinterpret_cast<const bf16x16*>(
            vp + (size_t)(nn * 16 + l15) * kNK + k2 * 32 + hf * 16);
        c = wmma_bf16(ap, bv, c);
      }
      #pragma unroll
      for (int r = 0; r < 8; ++r) {
        int m = w * 16 + r + 8 * hf;
        int d = nn * 16 + l15;
        s_ao[m * kC + h * kHD + d] = f2bf(c[r]);
      }
    }
  }
  __syncthreads();   // s_ao complete for all heads

  // ---- output projection: s_ao(64x192 bf16) @ pWt^T + pb -> f32 ----------
  f32x8 pacc[12];
  #pragma unroll
  for (int n = 0; n < 12; ++n)
    #pragma unroll
    for (int s = 0; s < 8; ++s) pacc[n][s] = 0.f;

  #pragma unroll
  for (int kb = 0; kb < kC; kb += 32) {
    bf16x16 af = *reinterpret_cast<const bf16x16*>(
        s_ao + (w * 16 + l15) * kC + kb + hf * 16);
    #pragma unroll
    for (int n = 0; n < 12; ++n) {
      bf16x16 bfrag = *reinterpret_cast<const bf16x16*>(
          pWt + (size_t)(n * 16 + l15) * kC + kb + hf * 16);
      pacc[n] = wmma_bf16(af, bfrag, pacc[n]);
    }
  }

  #pragma unroll
  for (int n = 0; n < 12; ++n) {
    int col = n * 16 + l15;
    float bb = pb[col];
    #pragma unroll
    for (int r = 0; r < 8; ++r) {
      int row = q0 + w * 16 + r + 8 * hf;          // row within window
      out[((size_t)b * kN + row) * kC + col] = pacc[n][r] + bb;
    }
  }
}

// ---------------------------------------------------------------------------
// Launcher. Workspace layout (~127 MB, every region a multiple of 256 B):
//   qWt 73728 | kvWt 18432 | pWt 73728 | bias 786432
//   qbf 100663296 | kbf 12582912 | vT 12582912
// ---------------------------------------------------------------------------
extern "C" void kernel_launch(void* const* d_in, const int* in_sizes, int n_in,
                              void* d_out, int out_size, void* d_ws, size_t ws_size,
                              hipStream_t stream) {
  const float* x    = (const float*)d_in[0];
  const float* q_w  = (const float*)d_in[1];
  const float* q_b  = (const float*)d_in[2];
  const float* kv_w = (const float*)d_in[3];
  const float* kv_b = (const float*)d_in[4];
  const float* rpb  = (const float*)d_in[5];
  const float* p_w  = (const float*)d_in[6];
  const float* p_b  = (const float*)d_in[7];
  float* out = (float*)d_out;

  char* ws = (char*)d_ws;
  unsigned short* qWt  = (unsigned short*)ws;  ws += (size_t)kC * kC * 2;        // 73728
  unsigned short* kvWt = (unsigned short*)ws;  ws += (size_t)48 * kC * 2;        // 18432
  unsigned short* pWt  = (unsigned short*)ws;  ws += (size_t)kC * kC * 2;        // 73728
  float*          bias = (float*)ws;           ws += (size_t)kNH * kN * kNK * 4; // 786432
  unsigned short* qbf  = (unsigned short*)ws;  ws += (size_t)kB * kN * kC * 2;   // 100663296
  unsigned short* kbf  = (unsigned short*)ws;  ws += (size_t)kB * kNH * kNK * kHD * 2;
  unsigned short* vT   = (unsigned short*)ws;  ws += (size_t)kB * kNH * kHD * kNK * 2;

  const int prepN = kC * kC + 48 * kC + kC * kC;          // 82944
  prep_weights<<<(prepN + 255) / 256, 256, 0, stream>>>(q_w, kv_w, p_w, qWt, kvWt, pWt);
  prep_bias<<<(kNH * kN * kNK + 255) / 256, 256, 0, stream>>>(rpb, bias);

  const int mBlocks = (kB * kN) / 128;                     // 2048
  gemm_qkv<<<mBlocks, 256, 0, stream>>>(x, qWt, kvWt, q_b, kv_b, qbf, kbf, vT);

  attn_proj<<<kB * 8, 128, 0, stream>>>(qbf, kbf, vT, bias, pWt, p_b, out);
}
